// WaveNet_45414984188359
// MI455X (gfx1250) — matrix operand order repaired and checked by reading, not compiled, not run on hardware
//
#include <hip/hip_runtime.h>
#include <hip/hip_bf16.h>

typedef __attribute__((ext_vector_type(16))) __bf16 v16bf;
typedef __attribute__((ext_vector_type(8)))  float  v8f;

#define WMMA_BF16(a, b, c) \
  __builtin_amdgcn_wmma_f32_16x16x32_bf16(false, (a), false, (b), (short)0, (c), false, false)

// ---- CDNA5 async memory->LDS copy (ASYNCcnt-tracked, no VGPR round trip) --
__device__ __forceinline__ uint32_t lds_off(const void* p) {
  // generic LDS address: low 32 bits are the LDS offset (ISA 10.2 aperture map)
  return (uint32_t)(uintptr_t)p;
}
__device__ __forceinline__ void async_b128(uint32_t dst_lds, const void* gsrc) {
  asm volatile("global_load_async_to_lds_b128 %0, %1, off"
               :: "v"(dst_lds), "v"(gsrc) : "memory");
}
__device__ __forceinline__ void wait_async0() {
  asm volatile("s_wait_asynccnt 0" ::: "memory");
}

// ---- fragment loaders (wave32, contiguous k order 16h+s in A and B) -------------
// A (16x32 bf16, M x K): lane<16 -> M=lane, K=16h..16h+15
__device__ __forceinline__ v16bf frag_a(const __bf16* W, int m0, int kp, int k0) {
  const int lane = threadIdx.x & 31;
  const int half = lane >> 4;
  const __bf16* row = W + (m0 + (lane & 15)) * kp + k0;
  v16bf a;
#pragma unroll
  for (int j = 0; j < 8; ++j) a[j] = row[16 * half + j];
#pragma unroll
  for (int j = 0; j < 8; ++j) a[8 + j] = row[16 * half + 8 + j];
  return a;
}
// B (32x16 bf16, K x N) staged in LDS as [N][Kpad]: lane n holds K=16h..16h+15
__device__ __forceinline__ v16bf frag_b(const __bf16* X, int n0, int kp, int k0) {
  const int lane = threadIdx.x & 31;
  const int half = lane >> 4;
  const __bf16* col = X + (n0 + (lane & 15)) * kp + k0 + 16 * half;
  v16bf b;
#pragma unroll
  for (int j = 0; j < 16; ++j) b[j] = col[j];
  return b;
}

// ===== start conv: res[b][t][c] (bf16) = Ws[32x256] * x[b][c][t] ==========
__global__ __launch_bounds__(128)
void wn_start(const float* __restrict__ x, const float* __restrict__ w,
              __bf16* __restrict__ res, int T, int tilesPerB) {
  const int C = 256, R = 32, NT = 64, KP = 264;
  __shared__ __bf16 Wl[32 * 264];
  __shared__ __bf16 Xl[64 * 264];
  const int tile = blockIdx.x % tilesPerB;
  const int b    = blockIdx.x / tilesPerB;
  const int t0   = tile * NT;
  const int tid  = threadIdx.x;

  for (int i = tid; i < R * C; i += 128) {
    int m = i >> 8, k = i & 255;
    Wl[m * KP + k] = (__bf16)w[m * C + k];
  }
  for (int i = tid; i < C * NT; i += 128) {
    int c = i >> 6, n = i & 63;
    Xl[n * KP + c] = (__bf16)x[(size_t)b * C * T + (size_t)c * T + t0 + n];
  }
  __syncthreads();

  const int wid = tid >> 5;
  const int n0  = wid * 16;
  v8f acc0 = {}, acc1 = {};
#pragma unroll
  for (int k0 = 0; k0 < 256; k0 += 32) {
    v16bf bf = frag_b(Xl, n0, KP, k0);
    acc0 = WMMA_BF16(frag_a(Wl,  0, KP, k0), bf, acc0);
    acc1 = WMMA_BF16(frag_a(Wl, 16, KP, k0), bf, acc1);
  }
  const int lane = tid & 31;
  const int half = lane >> 4;
  // transposed bf16 store: row of 32 channels is contiguous (2x16B per lane)
  __bf16* row = res + (size_t)b * T * 32 + (size_t)(t0 + n0 + (lane & 15)) * 32;
#pragma unroll
  for (int g = 0; g < 8; ++g) {
    row[8 * half + g]      = (__bf16)acc0[g];
    row[16 + 8 * half + g] = (__bf16)acc1[g];
  }
}

// ===== one WaveNet layer: dilated conv + gate + 1x1 + residual/skip =======
// residual buffers: bf16, layout [b][t][32]; skips: f32, layout [b][t][32]
__global__ __launch_bounds__(256)
void wn_layer(const __bf16* __restrict__ res_in, __bf16* __restrict__ res_out,
              float* __restrict__ skips,
              const float* __restrict__ dw,   // [32][32][2] this layer
              const float* __restrict__ pw,   // [32][32]    this layer
              int T, int tilesPerB, int dil, int firstLayer) {
  const int NT = 128, KP = 40;
  __shared__ __bf16 W0[32 * 40], W1[32 * 40], P[32 * 40];
  __shared__ __bf16 Xc[128 * 40], Xs[128 * 40], H[128 * 40];
  const int tile = blockIdx.x % tilesPerB;
  const int b    = blockIdx.x / tilesPerB;
  const int t0   = tile * NT;
  const int tid  = threadIdx.x;
  const size_t base = (size_t)b * T * 32;

  // async staging of the residual tile (raw bf16, contiguous rows of 64B)
  for (int i = tid; i < 512; i += 256) {
    int n = i >> 2, q = (i & 3) * 8;
    async_b128(lds_off(&Xc[n * KP + q]), res_in + base + (size_t)(t0 + n) * 32 + q);
    int ts = t0 + n - dil;
    if (ts >= 0) {
      async_b128(lds_off(&Xs[n * KP + q]), res_in + base + (size_t)ts * 32 + q);
    } else {
      const uint4 z4 = {0u, 0u, 0u, 0u};
      *(uint4*)&Xs[n * KP + q] = z4;   // causal zero padding
    }
  }
  for (int i = tid; i < 32 * 32; i += 256) {
    int m = i >> 5, k = i & 31;
    W0[m * KP + k] = (__bf16)dw[i * 2 + 0];
    W1[m * KP + k] = (__bf16)dw[i * 2 + 1];
    P [m * KP + k] = (__bf16)pw[i];
  }
  wait_async0();
  __syncthreads();

  const int wid = tid >> 5, lane = tid & 31;
  const int n0 = wid * 16;
  v8f h0 = {}, h1 = {};
  {
    v16bf bs = frag_b(Xs, n0, KP, 0);
    v16bf bc = frag_b(Xc, n0, KP, 0);
    h0 = WMMA_BF16(frag_a(W0,  0, KP, 0), bs, h0);
    h0 = WMMA_BF16(frag_a(W1,  0, KP, 0), bc, h0);
    h1 = WMMA_BF16(frag_a(W0, 16, KP, 0), bs, h1);
    h1 = WMMA_BF16(frag_a(W1, 16, KP, 0), bc, h1);
  }
  // gated activation unit, reshape D-layout -> [n][k] in LDS
  const int half = lane >> 4;
  const int nn   = n0 + (lane & 15);
#pragma unroll
  for (int g = 0; g < 8; ++g) {
    float v0 = h0[g], v1 = h1[g];
    v0 = tanhf(v0) * (1.f / (1.f + __expf(-v0)));
    v1 = tanhf(v1) * (1.f / (1.f + __expf(-v1)));
    H[nn * KP + (g + 8 * half)]      = (__bf16)v0;
    H[nn * KP + (16 + g + 8 * half)] = (__bf16)v1;
  }
  __syncthreads();

  v8f s0 = {}, s1 = {};
  {
    v16bf bh = frag_b(H, n0, KP, 0);
    s0 = WMMA_BF16(frag_a(P,  0, KP, 0), bh, s0);
    s1 = WMMA_BF16(frag_a(P, 16, KP, 0), bh, s1);
  }
  // contiguous updates: one 32-channel row per lane (two 8-elem chunks/acc)
  const __bf16* ri = res_in  + base + (size_t)(t0 + nn) * 32;
  __bf16*       ro = res_out + base + (size_t)(t0 + nn) * 32;
  float*        sk = skips   + base + (size_t)(t0 + nn) * 32;
#pragma unroll
  for (int g = 0; g < 8; ++g) {
    int m = 8 * half + g;
    float v = s0[g];
    ro[m] = (__bf16)((float)ri[m] + v);
    sk[m] = firstLayer ? v : (sk[m] + v);
    int m2 = 16 + 8 * half + g;
    float v2 = s1[g];
    ro[m2] = (__bf16)((float)ri[m2] + v2);
    sk[m2] = firstLayer ? v2 : (sk[m2] + v2);
  }
}

// ===== fused head: relu(skips) -> W1[256x32] -> relu -> W2[256x256] =======
__global__ __launch_bounds__(256)
void wn_end(const float* __restrict__ skips, const float* __restrict__ w1,
            const float* __restrict__ w2, float* __restrict__ out,
            int T, int tilesPerB) {
  const int NT = 32, KP = 40, KPH = 264;
  __shared__ __bf16 S  [32 * 40];    // relu(skips) tile, [n][c]
  __shared__ __bf16 W1l[256 * 40];   // [m][k]
  __shared__ __bf16 W2p[256 * 40];   // k-panel of end_w2, [m][kk]
  __shared__ __bf16 Hd [32 * 264];   // hidden, [n][k]  (k = 256)
  const int tile = blockIdx.x % tilesPerB;
  const int b    = blockIdx.x / tilesPerB;
  const int t0   = tile * NT;
  const int tid  = threadIdx.x;

  // skips tile is fully contiguous in [b][t][c] layout
  const float* sp = skips + (size_t)b * T * 32 + (size_t)t0 * 32;
  for (int i = tid; i < 32 * NT; i += 256) {
    S[(i >> 5) * KP + (i & 31)] = (__bf16)fmaxf(sp[i], 0.f);
  }
  for (int i = tid; i < 256 * 32; i += 256) {
    int m = i >> 5, k = i & 31;
    W1l[m * KP + k] = (__bf16)w1[i];
  }
  __syncthreads();

  const int wid = tid >> 5, lane = tid & 31;
  const int half = lane >> 4;
  // phase 1: hidden = relu(W1 * relu(S)); 32 tiles (16 m-tiles x 2 col-grps)
#pragma unroll
  for (int j = 0; j < 4; ++j) {
    int q = wid * 4 + j;
    int mt = q >> 1, cg = q & 1;
    v8f acc = {};
    acc = WMMA_BF16(frag_a(W1l, mt * 16, KP, 0), frag_b(S, cg * 16, KP, 0), acc);
    int n = cg * 16 + (lane & 15);
#pragma unroll
    for (int g = 0; g < 8; ++g) {
      int m = mt * 16 + g + 8 * half;
      Hd[n * KPH + m] = (__bf16)fmaxf(acc[g], 0.f);
    }
  }

  // phase 2: logits = W2 * hidden, streaming W2 as LDS k-panels
  v8f acc2[4] = {v8f{}, v8f{}, v8f{}, v8f{}};
  for (int k0 = 0; k0 < 256; k0 += 32) {
    __syncthreads();   // panel consumers done (k0==0: Hd writes visible)
    for (int i = tid; i < 256 * 32; i += 256) {
      int m = i >> 5, kk = i & 31;
      W2p[m * KP + kk] = (__bf16)w2[(size_t)m * 256 + k0 + kk];
      if (k0 + 32 < 256) __builtin_prefetch(&w2[(size_t)m * 256 + k0 + 32 + kk], 0, 1);
    }
    __syncthreads();
#pragma unroll
    for (int j = 0; j < 4; ++j) {
      int q = wid * 4 + j;
      int mt = q >> 1, cg = q & 1;
      acc2[j] = WMMA_BF16(frag_a(W2p, mt * 16, KP, 0),
                          frag_b(Hd, cg * 16, KPH, k0), acc2[j]);
    }
  }
  // required output layout [b][256][T] (f32)
  const size_t obase = (size_t)b * 256 * T;
#pragma unroll
  for (int j = 0; j < 4; ++j) {
    int q = wid * 4 + j;
    int mt = q >> 1, cg = q & 1;
    int n = t0 + cg * 16 + (lane & 15);
#pragma unroll
    for (int g = 0; g < 8; ++g) {
      int m = mt * 16 + g + 8 * half;
      out[obase + (size_t)m * T + n] = acc2[j][g];
    }
  }
}

extern "C" void kernel_launch(void* const* d_in, const int* in_sizes, int n_in,
                              void* d_out, int out_size, void* d_ws, size_t ws_size,
                              hipStream_t stream) {
  (void)in_sizes; (void)n_in; (void)out_size; (void)ws_size;
  const float* x   = (const float*)d_in[0];
  const float* sw  = (const float*)d_in[1];
  const float* dw  = (const float*)d_in[2];
  const float* pw  = (const float*)d_in[3];
  const float* ew1 = (const float*)d_in[4];
  const float* ew2 = (const float*)d_in[5];
  float* out = (float*)d_out;

  const int B = 8, T = 16000;
  const size_t resElems = (size_t)B * T * 32;          // 4.096M elements
  __bf16* resA = (__bf16*)d_ws;                        // 8.19 MB
  __bf16* resB = resA + resElems;                      // 8.19 MB
  float*  skp  = (float*)(resB + resElems);            // 16.38 MB (f32 accum)

  wn_start<<<dim3(B * (T / 64)), dim3(128), 0, stream>>>(x, sw, resA, T, T / 64);

  const __bf16* rin = resA;
  __bf16* rout = resB;
  int li = 0;
  for (int blk = 0; blk < 4; ++blk) {
    int dil = 1;
    for (int l = 0; l < 10; ++l) {
      wn_layer<<<dim3(B * (T / 128)), dim3(256), 0, stream>>>(
          rin, rout, skp, dw + (size_t)li * 2048, pw + (size_t)li * 1024,
          T, T / 128, dil, li == 0 ? 1 : 0);
      const __bf16* tmp = rout; rout = (__bf16*)rin; rin = tmp;
      dil <<= 1; ++li;
    }
  }

  wn_end<<<dim3(B * (T / 32)), dim3(256), 0, stream>>>(skp, ew1, ew2, out, T, T / 32);
}
